// GoSpecificWattiPooling_33294586478814
// MI455X (gfx1250) — matrix operand order repaired and checked by reading, not compiled, not run on hardware
//
#include <hip/hip_runtime.h>
#include <hip/hip_bf16.h>

typedef __bf16 bf16_t;
typedef __attribute__((ext_vector_type(16))) __bf16 v16bf;
typedef __attribute__((ext_vector_type(8)))  __bf16 v8bf;
typedef __attribute__((ext_vector_type(8)))  float  v8f;

#define B_  16
#define L_  8192
#define DH_ 256
#define T_  512
#define DG_ 512
#define DP_ 256

#define WMMA_BF16(a,b,c) \
  __builtin_amdgcn_wmma_f32_16x16x32_bf16(false,(a),false,(b),(short)0,(c),false,false)

static __device__ __forceinline__ int lane_id() { return (int)(threadIdx.x & 31); }

// A-operand (16x32 bf16): lanes 0-15 hold K=0..15 contiguous, lanes 16-31 hold K=16..31
static __device__ __forceinline__ v16bf load_a_bf16(const bf16_t* base, int stride) {
  const int lane = lane_id();
  const int r = lane & 15, h = lane >> 4;
  const bf16_t* p = base + (size_t)r * stride + h * 16;
  v8bf lo = *(const v8bf*)(p);
  v8bf hi = *(const v8bf*)(p + 8);
  return __builtin_shufflevector(lo, hi, 0,1,2,3,4,5,6,7,8,9,10,11,12,13,14,15);
}

// B-operand (32x16 bf16): lanes 0-15 hold K=0..15 contiguous, lanes 16-31 hold K=16..31
static __device__ __forceinline__ v16bf load_b_bf16(const bf16_t* base, int stride) {
  const int lane = lane_id();
  const int r = lane & 15, h = lane >> 4;
  return *(const v16bf*)(base + (size_t)r * stride + h * 16);
}

// B-operand built from f32 memory with on-the-fly bf16 conversion (for weights)
static __device__ __forceinline__ v16bf load_b_f32(const float* base, int stride) {
  const int lane = lane_id();
  const int r = lane & 15, h = lane >> 4;
  const float* p = base + (size_t)r * stride + h * 16;
  v16bf v;
#pragma unroll
  for (int j = 0; j < 16; ++j) v[j] = (bf16_t)p[j];
  return v;
}

// ---------------- Projection: Y(rows x 256) = bf16( X(rows x DIN) * W(256 x DIN)^T )
// WT variant additionally emits a pre-transposed bf16 copy of X in 64-row tiles:
// Xt[b][l/64][d][l%64]  (exactly the attention Z-phase B-operand layout)
template<int DIN, bool WT>
__global__ __launch_bounds__(256) void proj_kernel(const float* __restrict__ X,
                                                   const float* __restrict__ W,
                                                   bf16_t* __restrict__ Y,
                                                   bf16_t* __restrict__ Xt) {
  __shared__ __align__(16) bf16_t sA[16 * DIN];
  const int tid = threadIdx.x;
  const size_t row0 = (size_t)blockIdx.x * 16;
  const float* Xblk = X + row0 * DIN;
#pragma unroll
  for (int i = 0; i < DIN / 64; ++i) {
    int idx = (i * 256 + tid) * 4;
    float4 f = *(const float4*)(Xblk + idx);
    sA[idx + 0] = (bf16_t)f.x; sA[idx + 1] = (bf16_t)f.y;
    sA[idx + 2] = (bf16_t)f.z; sA[idx + 3] = (bf16_t)f.w;
  }
  __syncthreads();

  if (WT) {  // transpose-scatter this 16-row slab into Xt (DIN == 256 here)
    const int d = tid;
    const size_t bidx = row0 >> 13;            // row0 / L_
    const size_t l    = row0 & (size_t)(L_ - 1);
    const size_t lt   = l >> 6;
    const unsigned l64 = (unsigned)(l & 63);
    __align__(16) bf16_t tmp[16];
#pragma unroll
    for (int r = 0; r < 16; ++r) tmp[r] = sA[r * DIN + d];
    bf16_t* dst = Xt + (((bidx * (L_ / 64) + lt) * DH_ + d) << 6) + l64;
    *(uint4*)(dst + 0) = *(const uint4*)(tmp + 0);
    *(uint4*)(dst + 8) = *(const uint4*)(tmp + 8);
  }

  const int w  = tid >> 5;
  const int c0 = w * 32;                 // 8 waves x 32 cols = 256 output cols
  v8f acc0 = {}; v8f acc1 = {};
#pragma unroll 4
  for (int kk = 0; kk < DIN; kk += 32) {
    v16bf a  = load_a_bf16(sA + kk, DIN);
    v16bf b0 = load_b_f32(W + (size_t)c0 * DIN + kk, DIN);
    v16bf b1 = load_b_f32(W + (size_t)(c0 + 16) * DIN + kk, DIN);
    acc0 = WMMA_BF16(a, b0, acc0);
    acc1 = WMMA_BF16(a, b1, acc1);
  }
  const int lane = lane_id();
  const int r16 = lane & 15, h = lane >> 4;
#pragma unroll
  for (int r = 0; r < 8; ++r) {
    size_t orow = row0 + r + h * 8;
    Y[orow * DP_ + c0 + r16]      = (bf16_t)acc0[r];
    Y[orow * DP_ + c0 + 16 + r16] = (bf16_t)acc1[r];
  }
}

// ---------------- Flash attention: one block = (b, 32 query rows), 8 waves, Ltile = 64
// H tile arrives via GLOBAL_LOAD_ASYNC_TO_LDS (ASYNCcnt), overlapped with S-WMMAs+softmax.
__global__ __launch_bounds__(256) void attn_kernel(const bf16_t* __restrict__ Qb,
                                                   const bf16_t* __restrict__ Kb,
                                                   const bf16_t* __restrict__ Hbt,
                                                   const unsigned char* __restrict__ mask,
                                                   float* __restrict__ Z) {
  __shared__ __align__(16) bf16_t sH[DH_ * 64];   // 32 KB, pre-transposed [d][l64]
  __shared__ __align__(16) float  sS[32 * 65];    // padded stride
  __shared__ __align__(16) bf16_t sP[32 * 72];    // padded stride (mult of 8)
  __shared__ float sM[32], sL[32], sR[32];

  const int tid  = threadIdx.x;
  const int w    = tid >> 5;
  const int lane = tid & 31;
  const int r16  = lane & 15, h = lane >> 4;
  const int b    = blockIdx.x >> 4;
  const int t0   = (blockIdx.x & 15) * 32;

  // Q A-operands in registers: wave w only ever multiplies rows rb_s*16..+15
  const int rb_s = w >> 2, cb_s = w & 3;
  v16bf qa[8];
  {
    const bf16_t* Qbase = Qb + ((size_t)b * T_ + t0 + rb_s * 16) * DP_;
#pragma unroll
    for (int kk = 0; kk < 8; ++kk) qa[kk] = load_a_bf16(Qbase + kk * 32, DP_);
  }
  if (tid < 32) { sM[tid] = -3.0e38f; sL[tid] = 0.0f; }

  v8f acc[4] = {};  // wave w owns tiles id=w*4+j: rows (id&1)*16, cols (id>>1)*16

  const bf16_t* Kbb  = Kb  + (size_t)b * L_ * DP_;
  const bf16_t* Hb_t = Hbt + (size_t)b * (L_ / 64) * (DH_ * 64);
  const unsigned char* mb = mask + (size_t)b * L_;
  const unsigned sH_lds = (unsigned)(uintptr_t)(void*)&sH[0];

  for (int l0 = 0; l0 < L_; l0 += 64) {
    __syncthreads();   // all waves finished reading sH from the previous tile

    { // kick off async copy of the 32 KB pre-transposed H tile (8 x b128 per lane)
      unsigned long long gbase =
          (unsigned long long)(uintptr_t)(Hb_t + (size_t)(l0 >> 6) * (DH_ * 64));
#pragma unroll
      for (int q = 0; q < 8; ++q) {
        unsigned boff = (unsigned)(w * 4096 + q * 512 + lane * 16);
        asm volatile("global_load_async_to_lds_b128 %0, %1, off"
                     :: "v"(sH_lds + boff), "v"(gbase + boff)
                     : "memory");
      }
    }

    { // S = Q * K^T (overlaps the async copy): wave w -> 16x16 tile
      v8f s = {};
#pragma unroll
      for (int kk = 0; kk < 8; ++kk) {
        v16bf bo = load_b_bf16(Kbb + (size_t)(l0 + cb_s * 16) * DP_ + kk * 32, DP_);
        s = WMMA_BF16(qa[kk], bo, s);
      }
      const int col = cb_s * 16 + r16;
      const bool mk = mb[l0 + col] != 0;
#pragma unroll
      for (int r = 0; r < 8; ++r) {
        float v = mk ? -1.0e38f : s[r] * 0.0625f;   // scale = DP^-0.5 = 1/16
        sS[(rb_s * 16 + r + h * 8) * 65 + col] = v;
      }
    }
    __syncthreads();

    // online softmax: one row per thread (still overlapped with async copy)
    if (tid < 32) {
      const int row = tid;
      float mOld = sM[row];
      float mNew = mOld;
#pragma unroll 8
      for (int c = 0; c < 64; ++c) mNew = fmaxf(mNew, sS[row * 65 + c]);
      float rfac, sum = 0.0f;
      if (mNew <= -9.0e37f) {   // fully masked so far
        rfac = 1.0f;
#pragma unroll 8
        for (int c = 0; c < 64; ++c) sP[row * 72 + c] = (bf16_t)0.0f;
      } else {
        rfac = __expf(mOld - mNew);
#pragma unroll 8
        for (int c = 0; c < 64; ++c) {
          float p = __expf(sS[row * 65 + c] - mNew);
          sP[row * 72 + c] = (bf16_t)p;
          sum += p;
        }
      }
      sM[row] = mNew;
      sL[row] = sL[row] * rfac + sum;
      sR[row] = rfac;
    }

    asm volatile("s_wait_asynccnt 0x0" ::: "memory");  // own tile slice landed in LDS
    __syncthreads();                                   // everyone's slice + sP visible

    // rescale accumulators, then Z += P * Htile
#pragma unroll
    for (int j = 0; j < 4; ++j) {
      const int id  = w * 4 + j;
      const int rb  = id & 1;
      const int col = (id >> 1) * 16;
#pragma unroll
      for (int r = 0; r < 8; ++r) acc[j][r] *= sR[rb * 16 + r + h * 8];
#pragma unroll
      for (int kk = 0; kk < 64; kk += 32) {
        v16bf a  = load_a_bf16(sP + rb * 16 * 72 + kk, 72);
        v16bf bo = load_b_bf16(sH + col * 64 + kk, 64);
        acc[j] = WMMA_BF16(a, bo, acc[j]);
      }
    }
  }
  __syncthreads();

  // Z = acc / l
#pragma unroll
  for (int j = 0; j < 4; ++j) {
    const int id  = w * 4 + j;
    const int rb  = id & 1;
    const int col = (id >> 1) * 16;
#pragma unroll
    for (int r = 0; r < 8; ++r) {
      int rowl = rb * 16 + r + h * 8;
      float lsum = sL[rowl];
      float inv = lsum > 0.0f ? 1.0f / lsum : 0.0f;
      Z[((size_t)b * T_ + t0 + rowl) * DH_ + col + r16] = acc[j][r] * inv;
    }
  }
}

extern "C" void kernel_launch(void* const* d_in, const int* in_sizes, int n_in,
                              void* d_out, int out_size, void* d_ws, size_t ws_size,
                              hipStream_t stream) {
  (void)in_sizes; (void)n_in; (void)out_size; (void)ws_size;
  const float* H  = (const float*)d_in[0];
  const float* G  = (const float*)d_in[1];
  const unsigned char* mask = (const unsigned char*)d_in[2];
  const float* Wk = (const float*)d_in[3];
  const float* Wq = (const float*)d_in[4];
  float* Z = (float*)d_out;

  bf16_t* Kb  = (bf16_t*)d_ws;                        // (B, L, DP)        bf16, 64 MB
  bf16_t* Qbp = Kb  + (size_t)B_ * L_ * DP_;          // (B, T, DP)        bf16,  4 MB
  bf16_t* Hbt = Qbp + (size_t)B_ * T_ * DP_;          // (B, L/64, DH, 64) bf16, 64 MB

  proj_kernel<DH_, true ><<<(B_ * L_) / 16, 256, 0, stream>>>(H, Wk, Kb, Hbt);
  proj_kernel<DG_, false><<<(B_ * T_) / 16, 256, 0, stream>>>(G, Wq, Qbp, nullptr);
  attn_kernel<<<B_ * (T_ / 32), 256, 0, stream>>>(Qbp, Kb, Hbt, mask, Z);
}
